// _self_attention_layer_11793980194850
// MI455X (gfx1250) — matrix operand order repaired and checked by reading, not compiled, not run on hardware
//
#include <hip/hip_runtime.h>

// ---------------------------------------------------------------------------
// Causal self-attention for MI455X (gfx1250, wave32, WMMA).
//
// Attention matmuls (~17 GFLOP) run on v_wmma_f32_16x16x32_f16 (f32 accum).
// 8 waves/block share double-buffered LDS K / V^T tiles; staging now uses the
// CDNA5 async DMA path (global_load_async_to_lds_b128, ASYNCcnt) instead of a
// register bounce, plus global_prefetch_b8 two tiles ahead.
//
// WMMA register layouts (ISA 7.12.2, wave32):
//   A (16x32 f16): lane m=lane&15, h=lane>>4 holds row m, K = 16h..16h+15 (code's order, not ISA's)
//   B (32x16 f16): lane n=lane&15, h=lane>>4 holds col n, K = 16h..16h+15 (code's order, not ISA's)
//   C/D (16x16 f32): VGPR r holds row 8h+r, col n
// ---------------------------------------------------------------------------

typedef _Float16 v16h __attribute__((ext_vector_type(16)));
typedef _Float16 v8h  __attribute__((ext_vector_type(8)));
typedef float    v8f  __attribute__((ext_vector_type(8)));

#define WMMA_F32_F16(a, b, c) \
  __builtin_amdgcn_wmma_f32_16x16x32_f16(false, (a), false, (b), (short)0, (c), false, false)

static __device__ __forceinline__ v16h cat8(v8h lo, v8h hi) {
  return __builtin_shufflevector(lo, hi, 0,1,2,3,4,5,6,7,8,9,10,11,12,13,14,15);
}

// CDNA5 async DMA: copy 16B global -> LDS, tracked by ASYNCcnt (GV mode).
// lds_off = LDS byte offset (low 32 bits of a generic shared-memory address,
// per the LDS aperture rule).
static __device__ __forceinline__ void async_copy_b128(unsigned lds_off, const void* gaddr) {
  asm volatile("global_load_async_to_lds_b128 %0, %1, off"
               :: "v"(lds_off), "v"(gaddr) : "memory");
}
static __device__ __forceinline__ void wait_asynccnt0() {
  asm volatile("s_wait_asynccnt 0" ::: "memory");
}

#define BATCH 4
#define SEQ   4096
#define DIM   256
#define HDIM  64

// ---------------------------------------------------------------------------
// Kernel 1: QKV projection. X[B*S,256] f32 @ W[256,64] f32 -> f16 outputs.
// q,k row-major [B*S,64]; v stored TRANSPOSED as vt[B,64,S] so the attention
// kernel's P@V B-operand is a contiguous load.
// ---------------------------------------------------------------------------
__global__ __launch_bounds__(256, 1)
void qkv_proj_kernel(const float* __restrict__ X,
                     const float* __restrict__ Wq,
                     const float* __restrict__ Wk,
                     const float* __restrict__ Wv,
                     _Float16* __restrict__ q,
                     _Float16* __restrict__ k,
                     _Float16* __restrict__ vt) {
  // W transposed in LDS as f16, pitch 264 (=256+8) -> conflict-free column reads.
  __shared__ __align__(16) _Float16 WT[64 * 264];

  const int tid  = threadIdx.x;
  const int lane = tid & 31;
  const int wid  = tid >> 5;
  const int n = lane & 15;
  const int h = lane >> 4;
  const int rowBase = blockIdx.x * 128 + wid * 16;

  for (int w = 0; w < 3; ++w) {
    const float* W = (w == 0) ? Wq : ((w == 1) ? Wk : Wv);
    for (int it = 0; it < 64; ++it) {
      int idx = it * 256 + tid;          // [0, 16384)
      int kk = idx >> 6, nn = idx & 63;  // W[kk][nn]
      WT[nn * 264 + kk] = (_Float16)W[idx];
    }
    __syncthreads();

    v8f acc0 = {}, acc1 = {}, acc2 = {}, acc3 = {};
#pragma unroll
    for (int s = 0; s < 8; ++s) {  // K step of 32
      const float* xb = X + (size_t)(rowBase + n) * DIM + s * 32 + 16 * h;
      v8f xlo = *(const v8f*)xb;
      v8f xhi = *(const v8f*)(xb + 8);
      v8h alo, ahi;
#pragma unroll
      for (int i = 0; i < 8; ++i) { alo[i] = (_Float16)xlo[i]; ahi[i] = (_Float16)xhi[i]; }
      v16h a = cat8(alo, ahi);
#pragma unroll
      for (int t = 0; t < 4; ++t) {
        const _Float16* wb = &WT[(t * 16 + n) * 264 + s * 32 + 16 * h];
        v16h bmat = cat8(*(const v8h*)wb, *(const v8h*)(wb + 8));
        v8f& acc = (t == 0) ? acc0 : ((t == 1) ? acc1 : ((t == 2) ? acc2 : acc3));
        acc = WMMA_F32_F16(a, bmat, acc);
      }
    }

#pragma unroll
    for (int t = 0; t < 4; ++t) {
      v8f acc = (t == 0) ? acc0 : ((t == 1) ? acc1 : ((t == 2) ? acc2 : acc3));
#pragma unroll
      for (int r = 0; r < 8; ++r) {
        int g = rowBase + 8 * h + r;
        _Float16 val = (_Float16)acc[r];
        if (w == 0)      q[(size_t)g * HDIM + t * 16 + n] = val;
        else if (w == 1) k[(size_t)g * HDIM + t * 16 + n] = val;
        else {
          int bb = g >> 12, ss = g & (SEQ - 1);
          vt[((size_t)bb * HDIM + t * 16 + n) * SEQ + ss] = val;
        }
      }
    }
    __syncthreads();
  }
}

// ---------------------------------------------------------------------------
// Kernel 2: flash attention (causal). Block = 128 query rows (8 waves x 16)
// of one batch. Per 32-key tile: K and V^T tiles DMA'd into double-buffered
// LDS (async-to-LDS) shared by all 8 waves; each wave: 4 WMMAs for S = Q K^T,
// online softmax, P through LDS (C->A relayout), 4 WMMAs for PV.
// ---------------------------------------------------------------------------
__global__ __launch_bounds__(256, 1)
void flash_attn_kernel(const _Float16* __restrict__ q,
                       const _Float16* __restrict__ k,
                       const _Float16* __restrict__ vt,
                       float* __restrict__ out) {
  __shared__ __align__(16) _Float16 Klds[2][32 * 72];  // 32 keys x 64 H, pitch 72
  __shared__ __align__(16) _Float16 Vlds[2][64 * 40];  // 64 H x 32 keys, pitch 40
  __shared__ __align__(16) _Float16 Pb[8][16 * 40];    // per-wave P tile, pitch 40

  const int tid  = threadIdx.x;
  const int lane = tid & 31;
  const int wid  = tid >> 5;
  const int n = lane & 15;
  const int h = lane >> 4;

  const int b  = blockIdx.x >> 5;   // 32 blocks per batch
  const int bi = blockIdx.x & 31;
  const int q0 = bi * 128 + wid * 16;

  const _Float16* qb = q  + (size_t)b * SEQ * HDIM;
  const _Float16* kb = k  + (size_t)b * SEQ * HDIM;
  const _Float16* vb = vt + (size_t)b * HDIM * SEQ;
  _Float16* Pw = &Pb[wid][0];

  // Cooperative staging map: one b128 chunk per thread per matrix per tile.
  const int kr = tid >> 3, kc = (tid & 7) * 8;  // K tile: row 0..31, col chunk
  const int vr = tid >> 2, vc = (tid & 3) * 8;  // V tile: H row 0..63, key chunk
  const unsigned kofs[2] = { (unsigned)(size_t)&Klds[0][kr * 72 + kc],
                             (unsigned)(size_t)&Klds[1][kr * 72 + kc] };
  const unsigned vofs[2] = { (unsigned)(size_t)&Vlds[0][vr * 40 + vc],
                             (unsigned)(size_t)&Vlds[1][vr * 40 + vc] };

  // Q tile in A layout, both H-halves
  v16h aq0, aq1;
  {
    const _Float16* p = qb + (size_t)(q0 + n) * HDIM + 16 * h;
    aq0 = cat8(*(const v8h*)p,        *(const v8h*)(p + 8));
    aq1 = cat8(*(const v8h*)(p + 32), *(const v8h*)(p + 40));
  }

  v8f o0 = {}, o1 = {}, o2 = {}, o3 = {};
  float mrow[8], lrow[8];
#pragma unroll
  for (int r = 0; r < 8; ++r) { mrow[r] = -1e30f; lrow[r] = 0.0f; }

  const int ntiles = 4 * (bi + 1);  // key tiles covering rows up to bi*128+127

  // DMA tile 0 into buffer 0.
  async_copy_b128(kofs[0], kb + (size_t)kr * HDIM + kc);
  async_copy_b128(vofs[0], vb + (size_t)vr * SEQ + vc);
  wait_asynccnt0();
  __syncthreads();

  for (int j = 0; j < ntiles; ++j) {
    const int key0 = j * 32;
    const int cur = j & 1, nxt = cur ^ 1;

    // Kick off next tile's DMA into the alternate buffer (overlaps compute;
    // WAR on that buffer was closed by the previous barrier).
    const bool haveNext = (j + 1 < ntiles);   // uniform across block
    if (haveNext) {
      async_copy_b128(kofs[nxt], kb + (size_t)(key0 + 32 + kr) * HDIM + kc);
      async_copy_b128(vofs[nxt], vb + (size_t)vr * SEQ + key0 + 32 + vc);
    }
    if (j + 2 < ntiles) {  // pull tile j+2 into GL2 (global_prefetch_b8)
      __builtin_prefetch(kb + (size_t)(key0 + 64 + kr) * HDIM + kc, 0, 1);
      __builtin_prefetch(vb + (size_t)vr * SEQ + key0 + 64 + vc, 0, 1);
    }

    // ---- scores: two 16x16 C tiles over 32 keys, K-dim = H = 64 ----
    v8f s0 = {}, s1 = {};
    {
      const _Float16* p0 = &Klds[cur][n * 72 + 16 * h];
      v16h bk00 = cat8(*(const v8h*)p0,        *(const v8h*)(p0 + 8));
      v16h bk01 = cat8(*(const v8h*)(p0 + 32), *(const v8h*)(p0 + 40));
      s0 = WMMA_F32_F16(aq0, bk00, s0);
      s0 = WMMA_F32_F16(aq1, bk01, s0);
      const _Float16* p1 = &Klds[cur][(16 + n) * 72 + 16 * h];
      v16h bk10 = cat8(*(const v8h*)p1,        *(const v8h*)(p1 + 8));
      v16h bk11 = cat8(*(const v8h*)(p1 + 32), *(const v8h*)(p1 + 40));
      s1 = WMMA_F32_F16(aq0, bk10, s1);
      s1 = WMMA_F32_F16(aq1, bk11, s1);
    }

    // ---- scale, causal mask, online softmax ----
#pragma unroll
    for (int r = 0; r < 8; ++r) {
      const int row = q0 + 8 * h + r;
      float x0 = s0[r] * 0.0625f;   // 1/sqrt(embedding_dim=256)
      float x1 = s1[r] * 0.0625f;
      if (key0 + n > row)      x0 = -1e9f;
      if (key0 + 16 + n > row) x1 = -1e9f;
      float tmax = fmaxf(x0, x1);
#pragma unroll
      for (int off = 1; off < 16; off <<= 1)
        tmax = fmaxf(tmax, __shfl_xor(tmax, off, 32));
      const float mnew = fmaxf(mrow[r], tmax);
      const float cf = __expf(mrow[r] - mnew);
      x0 = __expf(x0 - mnew);
      x1 = __expf(x1 - mnew);
      float sum = x0 + x1;
#pragma unroll
      for (int off = 1; off < 16; off <<= 1)
        sum += __shfl_xor(sum, off, 32);
      mrow[r] = mnew;
      lrow[r] = lrow[r] * cf + sum;
      o0[r] *= cf; o1[r] *= cf; o2[r] *= cf; o3[r] *= cf;
      Pw[(8 * h + r) * 40 + n]      = (_Float16)x0;
      Pw[(8 * h + r) * 40 + 16 + n] = (_Float16)x1;
    }

    // wave-internal DS store -> load ordering (CDNA5 split counter)
    asm volatile("s_wait_dscnt 0" ::: "memory");

    // ---- P in A layout ----
    v16h ap;
    {
      const _Float16* pp = &Pw[n * 40 + 16 * h];
      ap = cat8(*(const v8h*)pp, *(const v8h*)(pp + 8));
    }

    // ---- O += P (16x32) @ V (32x64) from LDS V^T ----
#pragma unroll
    for (int t = 0; t < 4; ++t) {
      const _Float16* vp = &Vlds[cur][(t * 16 + n) * 40 + 16 * h];
      v16h bv = cat8(*(const v8h*)vp, *(const v8h*)(vp + 8));
      v8f& o = (t == 0) ? o0 : ((t == 1) ? o1 : ((t == 2) ? o2 : o3));
      o = WMMA_F32_F16(ap, bv, o);
    }

    // ---- close out this iteration: DMA done, then block-wide barrier ----
    if (haveNext) wait_asynccnt0();
    __syncthreads();
  }

  // ---- epilogue: normalize, store f32 output [B,S,64] ----
#pragma unroll
  for (int r = 0; r < 8; ++r) {
    const float inv = 1.0f / lrow[r];
    const size_t g = ((size_t)b * SEQ + q0 + 8 * h + r) * HDIM;
    out[g +  0 + n] = o0[r] * inv;
    out[g + 16 + n] = o1[r] * inv;
    out[g + 32 + n] = o2[r] * inv;
    out[g + 48 + n] = o3[r] * inv;
  }
}

// ---------------------------------------------------------------------------
// Launch: ws holds q (2MB) | k (2MB) | v^T (2MB), all f16.
// ---------------------------------------------------------------------------
extern "C" void kernel_launch(void* const* d_in, const int* in_sizes, int n_in,
                              void* d_out, int out_size, void* d_ws, size_t ws_size,
                              hipStream_t stream) {
  const float* X  = (const float*)d_in[0];
  const float* Wq = (const float*)d_in[1];
  const float* Wk = (const float*)d_in[2];
  const float* Wv = (const float*)d_in[3];
  float* out = (float*)d_out;

  _Float16* ws = (_Float16*)d_ws;
  const size_t qkv_elems = (size_t)BATCH * SEQ * HDIM;  // 1,048,576
  _Float16* q  = ws;
  _Float16* k  = ws + qkv_elems;
  _Float16* vt = ws + 2 * qkv_elems;

  // 16384 rows / 128 rows per block
  qkv_proj_kernel<<<128, 256, 0, stream>>>(X, Wq, Wk, Wv, q, k, vt);
  // 4 batches x 32 row-blocks, 8 waves each
  flash_attn_kernel<<<128, 256, 0, stream>>>(q, k, vt, out);
}
